// SpatialEncoder_68281390071811
// MI455X (gfx1250) — matrix operand order repaired and checked by reading, not compiled, not run on hardware
//
#include <hip/hip_runtime.h>
#include <hip/hip_bf16.h>

// ---------------- problem constants (match reference) ----------------
static constexpr int BATCH = 4;
static constexpr int NNODE = 10000;
static constexpr int NEDGE = 160000;
static constexpr int E2    = NEDGE + NNODE;   // + self loops
static constexpr int INDIM = 128;
static constexpr int OUTD  = 256;             // HEADS*C
static constexpr int HEADS = 4;
static constexpr int CH    = 64;
static constexpr float NEG_SLOPE = 0.2f;

typedef __attribute__((ext_vector_type(16))) __bf16 v16bf;
typedef __attribute__((ext_vector_type(8)))  __bf16 v8bf;
typedef __attribute__((ext_vector_type(8)))  float  v8f;

// ---------------- helpers ----------------
__device__ __forceinline__ void atomicMaxF(float* addr, float v) {
  if (v >= 0.0f) atomicMax(reinterpret_cast<int*>(addr), __float_as_int(v));
  else           atomicMin(reinterpret_cast<unsigned int*>(addr), __float_as_uint(v));
}

// ---------------- prep: weight transpose+convert, x convert ----------------
__global__ void k_conv_w(const float* __restrict__ W, __hip_bfloat16* __restrict__ Wt,
                         int K, int O) {
  int id = blockIdx.x * blockDim.x + threadIdx.x;
  if (id >= K * O) return;
  int o = id / K, k = id % K;
  Wt[(size_t)o * K + k] = __float2bfloat16(W[(size_t)k * O + o]);
}

__global__ void k_conv_f32_bf16(const float* __restrict__ src,
                                __hip_bfloat16* __restrict__ dst, int n) {
  int id = blockIdx.x * blockDim.x + threadIdx.x;
  if (id < n) dst[id] = __float2bfloat16(src[id]);
}

// ---------------- GEMM: C[b][N][256] = A[b][N][K](bf16) * Wt[256][K](bf16) ----------------
// block = 128 threads (4 waves); wave w covers output cols [w*64, w*64+64)
// grid = (N/16, 1, B)
__global__ __launch_bounds__(128)
void k_gemm_wmma(const __hip_bfloat16* __restrict__ Abf,
                 const __hip_bfloat16* __restrict__ Wtbf,
                 float* __restrict__ C, int n_nodes, int K) {
  const int wave = threadIdx.x >> 5;
  const int lane = threadIdx.x & 31;
  const int half = lane >> 4;
  const int l16  = lane & 15;
  const int row0 = blockIdx.x * 16;
  const int b    = blockIdx.z;

  const __bf16* A  = reinterpret_cast<const __bf16*>(Abf) + (size_t)b * n_nodes * K;
  const __bf16* Wt = reinterpret_cast<const __bf16*>(Wtbf);

  v8f acc0 = {}, acc1 = {}, acc2 = {}, acc3 = {};
  const size_t arow = (size_t)(row0 + l16) * K;

  for (int k0 = 0; k0 < K; k0 += 32) {
    // A fragment (contiguous K order, matches B): elem e -> K = k0 + 16*half + e
    v8bf lo = *reinterpret_cast<const v8bf*>(A + arow + k0 + 16 * half);
    v8bf hi = *reinterpret_cast<const v8bf*>(A + arow + k0 + 16 * half + 8);
    v16bf af = __builtin_shufflevector(lo, hi, 0,1,2,3,4,5,6,7,8,9,10,11,12,13,14,15);

    // B fragments: lane holds column n = l16, 16 contiguous K values at k0 + 16*half
    const size_t koff = (size_t)k0 + 16 * half;
    v16bf b0 = *reinterpret_cast<const v16bf*>(Wt + (size_t)(wave*64 +  0 + l16) * K + koff);
    v16bf b1 = *reinterpret_cast<const v16bf*>(Wt + (size_t)(wave*64 + 16 + l16) * K + koff);
    v16bf b2 = *reinterpret_cast<const v16bf*>(Wt + (size_t)(wave*64 + 32 + l16) * K + koff);
    v16bf b3 = *reinterpret_cast<const v16bf*>(Wt + (size_t)(wave*64 + 48 + l16) * K + koff);

    acc0 = __builtin_amdgcn_wmma_f32_16x16x32_bf16(false, af, false, b0, (short)0, acc0, false, false);
    acc1 = __builtin_amdgcn_wmma_f32_16x16x32_bf16(false, af, false, b1, (short)0, acc1, false, false);
    acc2 = __builtin_amdgcn_wmma_f32_16x16x32_bf16(false, af, false, b2, (short)0, acc2, false, false);
    acc3 = __builtin_amdgcn_wmma_f32_16x16x32_bf16(false, af, false, b3, (short)0, acc3, false, false);
  }

  // C/D layout: VGPR r, lane l -> M = r + 8*half, N = l16
  float* Cb = C + ((size_t)b * n_nodes + row0) * OUTD;
  #pragma unroll
  for (int r = 0; r < 8; ++r) {
    size_t rowoff = (size_t)(r + 8 * half) * OUTD + wave * 64 + l16;
    Cb[rowoff +  0] = acc0[r];
    Cb[rowoff + 16] = acc1[r];
    Cb[rowoff + 32] = acc2[r];
    Cb[rowoff + 48] = acc3[r];
  }
}

// ---------------- attention logits + softmax-state init ----------------
// thread per (b, n, h)
__global__ void k_attn(const float* __restrict__ hW,
                       const float* __restrict__ a_src, const float* __restrict__ a_dst,
                       float* __restrict__ als, float* __restrict__ ald,
                       float* __restrict__ mx, float* __restrict__ den, int n_nodes) {
  int id = blockIdx.x * blockDim.x + threadIdx.x;
  int total = BATCH * n_nodes * HEADS;
  if (id >= total) return;
  int h = id % HEADS;
  int bn = id / HEADS;            // b*N + n
  const float* hw = hW + (size_t)bn * OUTD + h * CH;
  float s1 = 0.f, s2 = 0.f;
  #pragma unroll 8
  for (int c = 0; c < CH; ++c) {
    float v = hw[c];
    s1 += v * a_src[h * CH + c];
    s2 += v * a_dst[h * CH + c];
  }
  als[id] = s1;
  ald[id] = s2;
  mx[id]  = -3.0e38f;
  den[id] = 0.0f;
}

// ---------------- init accumulator with bias ----------------
__global__ void k_init_acc(float* __restrict__ acc, const float* __restrict__ bias, int total) {
  int id = blockIdx.x * blockDim.x + threadIdx.x;
  if (id < total) acc[id] = bias[id & (OUTD - 1)];
}

// ---------------- edge pass 1: segment max ----------------
// grid.x covers E2*HEADS, grid.y = batch
__global__ void k_edge_max(const int* __restrict__ ei,
                           const float* __restrict__ als, const float* __restrict__ ald,
                           float* __restrict__ mx, int n_nodes) {
  int id = blockIdx.x * blockDim.x + threadIdx.x;
  if (id >= E2 * HEADS) return;
  int e = id >> 2, h = id & 3;
  int b = blockIdx.y;
  int src, dst;
  if (e < NEDGE) { src = ei[e]; dst = ei[NEDGE + e]; } else { src = dst = e - NEDGE; }
  float v = als[((size_t)b * n_nodes + src) * HEADS + h] +
            ald[((size_t)b * n_nodes + dst) * HEADS + h];
  v = v > 0.f ? v : NEG_SLOPE * v;
  atomicMaxF(&mx[((size_t)b * n_nodes + dst) * HEADS + h], v);
}

// ---------------- edge pass 2: exp + denom ----------------
__global__ void k_edge_exp(const int* __restrict__ ei,
                           const float* __restrict__ als, const float* __restrict__ ald,
                           const float* __restrict__ mx,
                           float* __restrict__ exv, float* __restrict__ den, int n_nodes) {
  int id = blockIdx.x * blockDim.x + threadIdx.x;
  if (id >= E2 * HEADS) return;
  int e = id >> 2, h = id & 3;
  int b = blockIdx.y;
  int src, dst;
  if (e < NEDGE) { src = ei[e]; dst = ei[NEDGE + e]; } else { src = dst = e - NEDGE; }
  float v = als[((size_t)b * n_nodes + src) * HEADS + h] +
            ald[((size_t)b * n_nodes + dst) * HEADS + h];
  v = v > 0.f ? v : NEG_SLOPE * v;
  float ex = __expf(v - mx[((size_t)b * n_nodes + dst) * HEADS + h]);
  exv[((size_t)b * E2 + e) * HEADS + h] = ex;
  atomicAdd(&den[((size_t)b * n_nodes + dst) * HEADS + h], ex);
}

// ---------------- edge pass 3: weighted aggregation ----------------
// one block (256 threads) per edge: coalesced gather of hW[src], coalesced fp32 atomics to acc[dst]
__global__ __launch_bounds__(256)
void k_edge_aggr(const int* __restrict__ ei, const float* __restrict__ exv,
                 const float* __restrict__ den, const float* __restrict__ hW,
                 float* __restrict__ acc, int n_nodes) {
  int e = blockIdx.x;
  int b = blockIdx.y;
  int t = threadIdx.x;            // 0..255 ; head = t>>6
  int src, dst;
  if (e < NEDGE) { src = ei[e]; dst = ei[NEDGE + e]; } else { src = dst = e - NEDGE; }
  int h = t >> 6;
  float alpha = exv[((size_t)b * E2 + e) * HEADS + h] /
                (den[((size_t)b * n_nodes + dst) * HEADS + h] + 1e-16f);
  float contrib = alpha * hW[((size_t)b * n_nodes + src) * OUTD + t];
  atomicAdd(&acc[((size_t)b * n_nodes + dst) * OUTD + t], contrib);
}

// ---------------- ELU + (next-layer bf16 input | final fp32 output) ----------------
__global__ void k_finalize(const float* __restrict__ acc, __hip_bfloat16* __restrict__ hbf,
                           float* __restrict__ out, int total, int last) {
  int id = blockIdx.x * blockDim.x + threadIdx.x;
  if (id >= total) return;
  float v = acc[id];
  v = v > 0.f ? v : (__expf(v) - 1.0f);
  if (last) out[id] = v;
  else      hbf[id] = __float2bfloat16(v);
}

// ---------------- host launch ----------------
extern "C" void kernel_launch(void* const* d_in, const int* in_sizes, int n_in,
                              void* d_out, int out_size, void* d_ws, size_t ws_size,
                              hipStream_t stream) {
  (void)in_sizes; (void)n_in; (void)out_size; (void)ws_size;

  const float* x      = (const float*)d_in[0];
  const int*   ei     = (const int*)  d_in[1];   // edge_index [2][E]
  const float* W1     = (const float*)d_in[2];
  const float* asrc1  = (const float*)d_in[3];
  const float* adst1  = (const float*)d_in[4];
  const float* b1     = (const float*)d_in[5];
  const float* W2     = (const float*)d_in[6];
  const float* asrc2  = (const float*)d_in[7];
  const float* adst2  = (const float*)d_in[8];
  const float* b2     = (const float*)d_in[9];
  float* out = (float*)d_out;

  // workspace carve-up (256B aligned)
  char* p = (char*)d_ws;
  auto carve = [&](size_t bytes) { char* r = p; p += (bytes + 255) & ~(size_t)255; return r; };
  __hip_bfloat16* hbf = (__hip_bfloat16*)carve((size_t)BATCH * NNODE * OUTD * 2);
  float* hW  = (float*)carve((size_t)BATCH * NNODE * OUTD * 4);
  float* acc = (float*)carve((size_t)BATCH * NNODE * OUTD * 4);
  float* als = (float*)carve((size_t)BATCH * NNODE * HEADS * 4);
  float* ald = (float*)carve((size_t)BATCH * NNODE * HEADS * 4);
  float* mx  = (float*)carve((size_t)BATCH * NNODE * HEADS * 4);
  float* den = (float*)carve((size_t)BATCH * NNODE * HEADS * 4);
  float* exv = (float*)carve((size_t)BATCH * E2 * HEADS * 4);
  __hip_bfloat16* W1t = (__hip_bfloat16*)carve((size_t)OUTD * INDIM * 2);
  __hip_bfloat16* W2t = (__hip_bfloat16*)carve((size_t)OUTD * OUTD * 2);

  // weights: transpose + bf16
  {
    int n1 = INDIM * OUTD, n2 = OUTD * OUTD;
    k_conv_w<<<(n1 + 255) / 256, 256, 0, stream>>>(W1, W1t, INDIM, OUTD);
    k_conv_w<<<(n2 + 255) / 256, 256, 0, stream>>>(W2, W2t, OUTD, OUTD);
  }
  // layer-1 input: x -> bf16, contiguous [B][N][128]
  {
    int nx = BATCH * NNODE * INDIM;
    k_conv_f32_bf16<<<(nx + 255) / 256, 256, 0, stream>>>(x, hbf, nx);
  }

  const int totalBNO = BATCH * NNODE * OUTD;
  const int totalBNH = BATCH * NNODE * HEADS;

  for (int L = 0; L < 2; ++L) {
    const int K = (L == 0) ? INDIM : OUTD;
    const __hip_bfloat16* Wt = (L == 0) ? W1t : W2t;
    const float* as = (L == 0) ? asrc1 : asrc2;
    const float* ad = (L == 0) ? adst1 : adst2;
    const float* bi = (L == 0) ? b1 : b2;

    k_gemm_wmma<<<dim3(NNODE / 16, 1, BATCH), 128, 0, stream>>>(hbf, Wt, hW, NNODE, K);

    k_attn<<<(totalBNH + 255) / 256, 256, 0, stream>>>(hW, as, ad, als, ald, mx, den, NNODE);

    k_init_acc<<<(totalBNO + 255) / 256, 256, 0, stream>>>(acc, bi, totalBNO);

    dim3 eg((E2 * HEADS + 255) / 256, BATCH);
    k_edge_max<<<eg, 256, 0, stream>>>(ei, als, ald, mx, NNODE);
    k_edge_exp<<<eg, 256, 0, stream>>>(ei, als, ald, mx, exv, den, NNODE);

    k_edge_aggr<<<dim3(E2, BATCH), 256, 0, stream>>>(ei, exv, den, hW, acc, NNODE);

    k_finalize<<<(totalBNO + 255) / 256, 256, 0, stream>>>(acc, hbf, out, totalBNO, L == 1);
  }
}
